// AttentionModel_39848706572887
// MI455X (gfx1250) — matrix operand order repaired and checked by reading, not compiled, not run on hardware
//
#include <hip/hip_runtime.h>
#include <stdint.h>

typedef __attribute__((ext_vector_type(16))) _Float16 v16h;
typedef __attribute__((ext_vector_type(8)))  _Float16 v8h;
typedef __attribute__((ext_vector_type(4)))  _Float16 v4h;
typedef __attribute__((ext_vector_type(8)))  float    v8f;

#define S_LEN 2048
#define EDIM  128
#define NC    64            // keys per tile
#define NT    (S_LEN / NC)  // 32 key tiles
#define SK    136           // K tile LDS row stride (halves): 128 + 8 pad
#define SV    72            // V^T tile LDS row stride (halves): 64 + 8 pad
#define SP    72            // P scratch row stride (halves): 64 + 8 pad
#define KEEP_THRESH 3865470566u   // floor(0.9 * 2^32)

__device__ __forceinline__ uint32_t hashu(uint32_t x) {
    x *= 2654435761u;
    x ^= x >> 13;
    x *= 0x9E3779B9u;
    return x;
}

// 16-byte async global->LDS copy (per-lane), tracked by ASYNCcnt.
__device__ __forceinline__ void async_ld16(const _Float16* g, const _Float16* l) {
    const uint32_t laddr = (uint32_t)(uintptr_t)l;   // LDS aperture: low 32 bits = LDS byte offset
    const uint64_t gaddr = (uint64_t)(uintptr_t)g;
    asm volatile("global_load_async_to_lds_b128 %0, %1, off"
                 :: "v"(laddr), "v"(gaddr) : "memory");
}

// ---------------- pre-pass: K f32 -> f16 (same layout) ----------------
__global__ __launch_bounds__(256)
void cvt_k_kernel(const float* __restrict__ K, _Float16* __restrict__ Kh) {
    const size_t i = ((size_t)blockIdx.x * 256 + threadIdx.x) * 4;
    const float4 f = *(const float4*)(K + i);
    v4h h; h[0] = (_Float16)f.x; h[1] = (_Float16)f.y;
           h[2] = (_Float16)f.z; h[3] = (_Float16)f.w;
    *(v4h*)(Kh + i) = h;
}

// ---------------- pre-pass: V f32 [b][n][e] -> f16 transposed [b][e][n] ----------------
__global__ __launch_bounds__(256)
void cvt_vt_kernel(const float* __restrict__ V, _Float16* __restrict__ Vt) {
    __shared__ float tile[32][33];
    const int b  = blockIdx.z;
    const int n0 = blockIdx.x * 32;
    const int e0 = blockIdx.y * 32;
    const int tx = threadIdx.x & 31;
    const int ty = threadIdx.x >> 5;   // 0..7
    #pragma unroll
    for (int i = 0; i < 4; ++i) {
        const int n = n0 + ty + i * 8;
        tile[ty + i * 8][tx] = V[((size_t)(b * S_LEN + n)) * EDIM + e0 + tx];
    }
    __syncthreads();
    #pragma unroll
    for (int i = 0; i < 4; ++i) {
        const int e = e0 + ty + i * 8;
        Vt[((size_t)(b * EDIM + e)) * S_LEN + n0 + tx] = (_Float16)tile[tx][ty + i * 8];
    }
}

// ---------------- main flash-attention kernel (f16 K / V^T in workspace) ----------------
__global__ __launch_bounds__(256, 1)
void fa_fwd_f16_kernel(const float* __restrict__ Q, const _Float16* __restrict__ Kh,
                       const _Float16* __restrict__ Vt, float* __restrict__ O) {
    __shared__ _Float16 Ksh[2][NC * SK];     // [n][e], double buffered
    __shared__ _Float16 Vsh[2][EDIM * SV];   // transposed [e][n], double buffered
    __shared__ _Float16 Psh[8 * 16 * SP];    // per-wave P scratch [wave][m][n]

    const int b     = blockIdx.y;
    const int qBase = blockIdx.x * 128;
    const int tid   = threadIdx.x;
    const int wave  = tid >> 5;
    const int lane  = tid & 31;
    const int l16   = lane & 15;
    const bool hi   = lane >= 16;

    const _Float16* kgb = Kh + ((size_t)b * S_LEN) * EDIM;
    const _Float16* vgb = Vt + ((size_t)b * EDIM) * S_LEN;

    // ---- Q fragments (A-layout, f16), loaded once: 16x128 per wave ----
    v16h qa[4];
    {
        const int q = qBase + wave * 16 + l16;
        const float* qp = Q + ((size_t)(b * S_LEN + q)) * EDIM;
        #pragma unroll
        for (int c = 0; c < 4; ++c) {
            const int base0 = c * 32 + (hi ? 8 : 0);
            #pragma unroll
            for (int e = 0; e < 8; ++e) {
                qa[c][e]     = (_Float16)qp[base0 + e];
                qa[c][e + 8] = (_Float16)qp[base0 + 16 + e];
            }
        }
    }

    v8f acc[8];
    #pragma unroll
    for (int t = 0; t < 8; ++t) { v8f z = {}; acc[t] = z; }
    float mrun[8], lrun[8];
    #pragma unroll
    for (int r = 0; r < 8; ++r) { mrun[r] = -3.0e38f; lrun[r] = 0.f; }

    // stage key-tile kt into buffer buf: 8 x 16B async copies per thread
    auto stage = [&](int kt, int buf) {
        const int kBase = kt * NC;
        const _Float16* kg = kgb + (size_t)kBase * EDIM;
        const _Float16* vg = vgb + kBase;
        #pragma unroll
        for (int i = 0; i < 4; ++i) {
            const int j = tid + i * 256;           // 0..1023
            const int krow = j >> 4, ks = j & 15;  // 64 rows x 16 segs
            async_ld16(kg + krow * EDIM + ks * 8, &Ksh[buf][krow * SK + ks * 8]);
            const int vrow = j >> 3, vs = j & 7;   // 128 rows x 8 segs
            async_ld16(vg + (size_t)vrow * S_LEN + vs * 8, &Vsh[buf][vrow * SV + vs * 8]);
        }
    };

    stage(0, 0);

    for (int kt = 0; kt < NT; ++kt) {
        const int kBase = kt * NC;
        const int buf = kt & 1;

        asm volatile("s_wait_asynccnt 0x0" ::: "memory");  // my copies into buf done
        __syncthreads();                                   // everyone's copies done
        if (kt + 1 < NT) stage(kt + 1, buf ^ 1);           // prefetch overlaps compute

        const _Float16* Kb = &Ksh[buf][0];
        const _Float16* Vb = &Vsh[buf][0];

        // ---- scores: S(16x64) = Q(16x128) * K_tile^T ----
        v8f sc[4];
        #pragma unroll
        for (int t = 0; t < 4; ++t) {
            v8f c = {};
            #pragma unroll
            for (int ch = 0; ch < 4; ++ch) {
                union { v16h v; v8h h[2]; } bu;
                const _Float16* kb = &Kb[(t * 16 + l16) * SK + ch * 32 + (hi ? 8 : 0)];
                bu.h[0] = *(const v8h*)kb;
                bu.h[1] = *(const v8h*)(kb + 16);
                c = __builtin_amdgcn_wmma_f32_16x16x32_f16(
                        false, qa[ch], false, bu.v, (short)0, c, false, false);
            }
            sc[t] = c;
        }

        // ---- online softmax: row max via half-wave shfl reduction ----
        float scale[8];
        #pragma unroll
        for (int r = 0; r < 8; ++r) {
            float mx = fmaxf(fmaxf(sc[0][r], sc[1][r]), fmaxf(sc[2][r], sc[3][r]));
            #pragma unroll
            for (int msk = 1; msk < 16; msk <<= 1)
                mx = fmaxf(mx, __shfl_xor(mx, msk, 32));
            const float mnew = fmaxf(mrun[r], mx);
            scale[r] = __expf(mrun[r] - mnew);
            mrun[r]  = mnew;
        }

        // ---- exp, rowsum (pre-dropout), dropout, write P (f16) to LDS ----
        float rsum[8];
        #pragma unroll
        for (int r = 0; r < 8; ++r) rsum[r] = 0.f;
        const int qg = qBase + wave * 16;
        const uint32_t idxA = ((uint32_t)(b * S_LEN + qg)) * (uint32_t)S_LEN
                              + (uint32_t)kBase + (uint32_t)l16;
        #pragma unroll
        for (int t = 0; t < 4; ++t) {
            #pragma unroll
            for (int r = 0; r < 8; ++r) {
                const float p = __expf(sc[t][r] - mrun[r]);
                rsum[r] += p;
                const int m = (hi ? 8 : 0) + r;
                const uint32_t idx = idxA + (uint32_t)m * (uint32_t)S_LEN + (uint32_t)(t * 16);
                const float pd = (hashu(idx) < KEEP_THRESH) ? p * (1.0f / 0.9f) : 0.f;
                Psh[(wave * 16 + m) * SP + t * 16 + l16] = (_Float16)pd;
            }
        }
        #pragma unroll
        for (int r = 0; r < 8; ++r) {
            float s = rsum[r];
            #pragma unroll
            for (int msk = 1; msk < 16; msk <<= 1)
                s += __shfl_xor(s, msk, 32);
            lrun[r] = lrun[r] * scale[r] + s;
            #pragma unroll
            for (int t = 0; t < 8; ++t)
                acc[t][r] *= scale[r];
        }

        // same-wave LDS is in-order; make the P write->read dependency explicit
        asm volatile("s_wait_dscnt 0" ::: "memory");

        // ---- P A-fragments from LDS scratch ----
        v16h pa[2];
        #pragma unroll
        for (int c = 0; c < 2; ++c) {
            const _Float16* pp = &Psh[(wave * 16 + l16) * SP + c * 32 + (hi ? 8 : 0)];
            union { v16h v; v8h h[2]; } u;
            u.h[0] = *(const v8h*)pp;
            u.h[1] = *(const v8h*)(pp + 16);
            pa[c] = u.v;
        }

        // ---- O(16x128) += P(16x64) * V_tile(64x128) ----
        #pragma unroll
        for (int t = 0; t < 8; ++t) {
            v8f c = acc[t];
            #pragma unroll
            for (int ch = 0; ch < 2; ++ch) {
                union { v16h v; v8h h[2]; } bu;
                const _Float16* vb = &Vb[(t * 16 + l16) * SV + ch * 32 + (hi ? 8 : 0)];
                bu.h[0] = *(const v8h*)vb;
                bu.h[1] = *(const v8h*)(vb + 16);
                c = __builtin_amdgcn_wmma_f32_16x16x32_f16(
                        false, pa[ch], false, bu.v, (short)0, c, false, false);
            }
            acc[t] = c;
        }

        __syncthreads();   // all reads of buf done before it is re-staged
    }

    // ---- normalize and store ----
    const int q0 = qBase + wave * 16;
    #pragma unroll
    for (int r = 0; r < 8; ++r) {
        const int m = (hi ? 8 : 0) + r;
        const float inv = 1.0f / lrun[r];
        float* op = O + ((size_t)(b * S_LEN + q0 + m)) * EDIM;
        #pragma unroll
        for (int t = 0; t < 8; ++t)
            op[t * 16 + l16] = acc[t][r] * inv;
    }
}

// ---------------- fallback: fully fused kernel (f32 inputs, in-kernel convert) ----------------
__global__ __launch_bounds__(256, 1)
void fa_fwd_kernel(const float* __restrict__ Q, const float* __restrict__ K,
                   const float* __restrict__ V, float* __restrict__ O) {
    __shared__ _Float16 Ksh[NC * SK];
    __shared__ _Float16 Vsh[EDIM * SV];
    __shared__ _Float16 Psh[8 * 16 * SP];

    const int b     = blockIdx.y;
    const int qBase = blockIdx.x * 128;
    const int tid   = threadIdx.x;
    const int wave  = tid >> 5;
    const int lane  = tid & 31;
    const int l16   = lane & 15;
    const bool hi   = lane >= 16;

    v16h qa[4];
    {
        const int q = qBase + wave * 16 + l16;
        const float* qp = Q + ((size_t)(b * S_LEN + q)) * EDIM;
        #pragma unroll
        for (int c = 0; c < 4; ++c) {
            const int base0 = c * 32 + (hi ? 8 : 0);
            #pragma unroll
            for (int e = 0; e < 8; ++e) {
                qa[c][e]     = (_Float16)qp[base0 + e];
                qa[c][e + 8] = (_Float16)qp[base0 + 16 + e];
            }
        }
    }

    v8f acc[8];
    #pragma unroll
    for (int t = 0; t < 8; ++t) { v8f z = {}; acc[t] = z; }
    float mrun[8], lrun[8];
    #pragma unroll
    for (int r = 0; r < 8; ++r) { mrun[r] = -3.0e38f; lrun[r] = 0.f; }

    for (int kt = 0; kt < NT; ++kt) {
        const int kBase = kt * NC;
        __syncthreads();
        {
            const float* kp = K + ((size_t)(b * S_LEN + kBase)) * EDIM;
            const float* vp = V + ((size_t)(b * S_LEN + kBase)) * EDIM;
            #pragma unroll
            for (int i = 0; i < 8; ++i) {
                const int j   = tid + i * 256;
                const int row = j >> 5;
                const int c4  = j & 31;
                float4 f = *(const float4*)(kp + row * EDIM + c4 * 4);
                v4h h; h[0] = (_Float16)f.x; h[1] = (_Float16)f.y;
                       h[2] = (_Float16)f.z; h[3] = (_Float16)f.w;
                *(v4h*)&Ksh[row * SK + c4 * 4] = h;
                float4 g = *(const float4*)(vp + row * EDIM + c4 * 4);
                Vsh[(c4 * 4 + 0) * SV + row] = (_Float16)g.x;
                Vsh[(c4 * 4 + 1) * SV + row] = (_Float16)g.y;
                Vsh[(c4 * 4 + 2) * SV + row] = (_Float16)g.z;
                Vsh[(c4 * 4 + 3) * SV + row] = (_Float16)g.w;
            }
        }
        __syncthreads();

        v8f sc[4];
        #pragma unroll
        for (int t = 0; t < 4; ++t) {
            v8f c = {};
            #pragma unroll
            for (int ch = 0; ch < 4; ++ch) {
                union { v16h v; v8h h[2]; } bu;
                const _Float16* kb = &Ksh[(t * 16 + l16) * SK + ch * 32 + (hi ? 8 : 0)];
                bu.h[0] = *(const v8h*)kb;
                bu.h[1] = *(const v8h*)(kb + 16);
                c = __builtin_amdgcn_wmma_f32_16x16x32_f16(
                        false, qa[ch], false, bu.v, (short)0, c, false, false);
            }
            sc[t] = c;
        }

        float scale[8];
        #pragma unroll
        for (int r = 0; r < 8; ++r) {
            float mx = fmaxf(fmaxf(sc[0][r], sc[1][r]), fmaxf(sc[2][r], sc[3][r]));
            #pragma unroll
            for (int msk = 1; msk < 16; msk <<= 1)
                mx = fmaxf(mx, __shfl_xor(mx, msk, 32));
            const float mnew = fmaxf(mrun[r], mx);
            scale[r] = __expf(mrun[r] - mnew);
            mrun[r]  = mnew;
        }

        float rsum[8];
        #pragma unroll
        for (int r = 0; r < 8; ++r) rsum[r] = 0.f;
        const int qg = qBase + wave * 16;
        #pragma unroll
        for (int t = 0; t < 4; ++t) {
            const int n = kBase + t * 16 + l16;
            #pragma unroll
            for (int r = 0; r < 8; ++r) {
                const float p = __expf(sc[t][r] - mrun[r]);
                rsum[r] += p;
                const int m = (hi ? 8 : 0) + r;
                const uint32_t idx = ((uint32_t)(b * S_LEN + (qg + m))) * (uint32_t)S_LEN + (uint32_t)n;
                const float pd = (hashu(idx) < KEEP_THRESH) ? p * (1.0f / 0.9f) : 0.f;
                Psh[(wave * 16 + m) * SP + t * 16 + l16] = (_Float16)pd;
            }
        }
        #pragma unroll
        for (int r = 0; r < 8; ++r) {
            float s = rsum[r];
            #pragma unroll
            for (int msk = 1; msk < 16; msk <<= 1)
                s += __shfl_xor(s, msk, 32);
            lrun[r] = lrun[r] * scale[r] + s;
            #pragma unroll
            for (int t = 0; t < 8; ++t)
                acc[t][r] *= scale[r];
        }

        asm volatile("s_wait_dscnt 0" ::: "memory");

        v16h pa[2];
        #pragma unroll
        for (int c = 0; c < 2; ++c) {
            const _Float16* pp = &Psh[(wave * 16 + l16) * SP + c * 32 + (hi ? 8 : 0)];
            union { v16h v; v8h h[2]; } u;
            u.h[0] = *(const v8h*)pp;
            u.h[1] = *(const v8h*)(pp + 16);
            pa[c] = u.v;
        }
        #pragma unroll
        for (int t = 0; t < 8; ++t) {
            v8f c = acc[t];
            #pragma unroll
            for (int ch = 0; ch < 2; ++ch) {
                union { v16h v; v8h h[2]; } bu;
                const _Float16* vb = &Vsh[(t * 16 + l16) * SV + ch * 32 + (hi ? 8 : 0)];
                bu.h[0] = *(const v8h*)vb;
                bu.h[1] = *(const v8h*)(vb + 16);
                c = __builtin_amdgcn_wmma_f32_16x16x32_f16(
                        false, pa[ch], false, bu.v, (short)0, c, false, false);
            }
            acc[t] = c;
        }
    }

    const int q0 = qBase + wave * 16;
    #pragma unroll
    for (int r = 0; r < 8; ++r) {
        const int m = (hi ? 8 : 0) + r;
        const float inv = 1.0f / lrun[r];
        float* op = O + ((size_t)(b * S_LEN + q0 + m)) * EDIM;
        #pragma unroll
        for (int t = 0; t < 8; ++t)
            op[t * 16 + l16] = acc[t][r] * inv;
    }
}

extern "C" void kernel_launch(void* const* d_in, const int* in_sizes, int n_in,
                              void* d_out, int out_size, void* d_ws, size_t ws_size,
                              hipStream_t stream) {
    (void)n_in; (void)out_size;
    const float* Qp = (const float*)d_in[0];
    const float* Kp = (const float*)d_in[1];
    const float* Vp = (const float*)d_in[2];
    float* Op = (float*)d_out;
    const int B = in_sizes[0] / (S_LEN * EDIM);
    const size_t elems = (size_t)B * S_LEN * EDIM;
    const size_t need  = elems * sizeof(_Float16) * 2;

    if (ws_size >= need) {
        _Float16* Kh = (_Float16*)d_ws;
        _Float16* Vt = Kh + elems;
        cvt_k_kernel<<<dim3((unsigned)(elems / 4 / 256)), dim3(256), 0, stream>>>(Kp, Kh);
        cvt_vt_kernel<<<dim3(S_LEN / 32, EDIM / 32, B), dim3(256), 0, stream>>>(Vp, Vt);
        fa_fwd_f16_kernel<<<dim3(S_LEN / 128, B), dim3(256), 0, stream>>>(Qp, Kh, Vt, Op);
    } else {
        fa_fwd_kernel<<<dim3(S_LEN / 128, B), dim3(256), 0, stream>>>(Qp, Kp, Vp, Op);
    }
}
